// LSTM_82867099009145
// MI455X (gfx1250) — matrix operand order repaired and checked by reading, not compiled, not run on hardware
//
#include <hip/hip_runtime.h>
#include <math.h>

typedef __attribute__((ext_vector_type(16))) __bf16 v16bf;
typedef __attribute__((ext_vector_type(8)))  __bf16 v8bf;
typedef __attribute__((ext_vector_type(4)))  __bf16 v4bf;
typedef __attribute__((ext_vector_type(8)))  float  v8f;

#define B_  64
#define T_  512
#define I_  1024
#define H_  1024
#define KC_ 2048          // I + H packed K extent
#define NBLK_ 64          // persistent grid: one block per 16 columns of H
#define PANEL_BYTES_ 262144ULL   // 64 k-chunks * 4 gates * 1024 B
#define RED_BYTES_   16384ULL    // 4 waves * 4 gates * 256 floats
#define LDS_BYTES_   (PANEL_BYTES_ + RED_BYTES_)

// ---- workspace layout (bytes) ----
#define WPK_OFF   0ULL
#define WPK_BYTES (NBLK_ * PANEL_BYTES_)                       // 16 MB
#define XBF_OFF   (WPK_OFF + WPK_BYTES)
#define XBF_BYTES ((unsigned long long)B_ * T_ * I_ * 2ULL)    // 64 MB
#define H0_OFF    (XBF_OFF + XBF_BYTES)
#define H_BYTES   ((unsigned long long)B_ * H_ * 2ULL)         // 128 KB
#define H1_OFF    (H0_OFF + H_BYTES)
#define CST_OFF   (H1_OFF + H_BYTES)
#define CST_BYTES ((unsigned long long)B_ * H_ * 4ULL)         // 256 KB
#define BAR_OFF   (CST_OFF + CST_BYTES)

static __device__ __forceinline__ float sigmoid_f(float x) {
  return 1.0f / (1.0f + __expf(-x));
}
static __device__ __forceinline__ float tanh_f(float x) {
  return 1.0f - 2.0f / (__expf(2.0f * x) + 1.0f);   // saturates to +-1
}

// Repack [w_ih | w_hh] (fp32) into bf16 WMMA-B fragment order, k-chunk major:
//   byte = nb*256KB + (kcg*4 + g)*1024 + half*512 + lane*16 + j*2
//   value = W[n = g*1024 + nb*16 + (lane&15)][K = kcg*32 + (lane>>4)*8 + half*16 + j]
// Within a k-chunk the 4 gates are adjacent (0/1024/2048/3072 B) -> DS
// immediate offsets; the two 16B halves at lane*16 (+512) tile all 64 banks.
__global__ void pack_weights(const float* __restrict__ w_ih,
                             const float* __restrict__ w_hh,
                             __bf16* __restrict__ wpack) {
  unsigned idx = blockIdx.x * blockDim.x + threadIdx.x;   // 8,388,608 elements
  unsigned j    = idx & 7;
  unsigned lane = (idx >> 3) & 31;
  unsigned half = (idx >> 8) & 1;
  unsigned g    = (idx >> 9) & 3;
  unsigned kcg  = (idx >> 11) & 63;
  unsigned nb   = idx >> 17;
  unsigned n = g * 1024u + nb * 16u + (lane & 15u);
  unsigned K = kcg * 32u + (lane >> 4) * 8u + half * 16u + j;
  float v = (K < (unsigned)I_) ? w_ih[(size_t)n * I_ + K]
                               : w_hh[(size_t)n * H_ + (K - I_)];
  wpack[idx] = (__bf16)v;
}

__global__ void convert_x(const float* __restrict__ x, __bf16* __restrict__ xb) {
  size_t i = (size_t)blockIdx.x * blockDim.x + threadIdx.x;
  const float4 v = ((const float4*)x)[i];
  v4bf o = { (__bf16)v.x, (__bf16)v.y, (__bf16)v.z, (__bf16)v.w };
  *(v4bf*)(xb + 4 * i) = o;
}

__global__ void init_state(__bf16* __restrict__ h0, __bf16* __restrict__ h1,
                           float* __restrict__ c, unsigned* __restrict__ bar) {
  int i = blockIdx.x * blockDim.x + threadIdx.x;   // B_*H_ threads
  h0[i] = (__bf16)0.0f;
  h1[i] = (__bf16)0.0f;
  c[i]  = 0.0f;
  if (i == 0) { bar[0] = 0u; bar[1] = 0u; }        // cnt, gen
}

// Persistent LSTM: one launch, T steps inside, weights LDS-resident.
// Grid 64 x 256 (8 waves). Block nb owns H-columns [nb*16, nb*16+16) for all
// 4 gates. Waves 0-3: x half (K 0..1023); waves 4-7: h half (K 1024..2047).
__global__ __launch_bounds__(256) void lstm_persistent(
    const __bf16* __restrict__ xb,     // [B][T][I] bf16
    const __bf16* __restrict__ wpack,  // fragment-ordered panels
    const float*  __restrict__ bias,   // [4096]
    __bf16* __restrict__ hbuf0, __bf16* __restrict__ hbuf1,
    float*  __restrict__ c_state,      // [B][H]
    float*  __restrict__ hs_out, float* __restrict__ cs_out,  // [B][T][H]
    unsigned* __restrict__ bar) {      // bar[0]=cnt, bar[1]=gen
  extern __shared__ __align__(32) char smem[];  // [0,256K): weights, then red

  const int tid  = threadIdx.x;
  const int lane = tid & 31;
  const int wave = tid >> 5;
  const int l16  = lane & 15;
  const int lhi  = lane >> 4;
  const int mt   = wave & 3;         // 16-row m-tile
  const int kh   = wave >> 2;        // 0: x-half, 1: h-half
  const int m0   = mt << 4;
  const int nb   = blockIdx.x;
  const int col  = (nb << 4) + l16;  // h column in [0,1024)

  // ---- one-time async preload of this block's 256 KB weight panel ----
  {
    const unsigned long long gbase =
        (unsigned long long)(uintptr_t)wpack + (unsigned long long)nb * PANEL_BYTES_;
    const unsigned lbase = (unsigned)(uintptr_t)smem;  // LDS byte offset
#pragma unroll 4
    for (int i = 0; i < 64; ++i) {                     // 64 * 4096 B = 256 KB
      unsigned off = ((unsigned)i << 12) + ((unsigned)tid << 4);
      unsigned ldst = lbase + off;
      unsigned long long gsrc = gbase + off;
      asm volatile("global_load_async_to_lds_b128 %0, %1, off"
                   :: "v"(ldst), "v"(gsrc) : "memory");
    }
    asm volatile("s_wait_asynccnt 0x0" ::: "memory");
  }
  __syncthreads();

  float bval[4];
#pragma unroll
  for (int g = 0; g < 4; ++g) bval[g] = bias[g * H_ + col];

  float* red = (float*)(smem + PANEL_BYTES_);  // [4][4][32][8] floats

  const size_t xrow = (size_t)(m0 + l16) * ((size_t)T_ * I_) + ((size_t)lhi << 3);
  const size_t hrow = (size_t)(m0 + l16) * H_ + ((size_t)lhi << 3);

  // prefetch assignment: thread covers row (tid>>2), 4 of 16 cachelines
  const int prow_i = tid >> 2;
  const int pcol_i = tid & 3;
  const char* pbase = (const char*)(xb + (size_t)prow_i * ((size_t)T_ * I_));

  for (int t = 0; t < T_; ++t) {
    const __bf16* hin  = (t & 1) ? hbuf1 : hbuf0;
    __bf16*       hout = (t & 1) ? hbuf0 : hbuf1;
    const __bf16* arow = (kh == 0) ? (xb + xrow + (size_t)t * I_) : (hin + hrow);

    // overlap: pull next timestep's x slice (128 KB, 1024 lines) into L2/L1
    if (t + 1 < T_) {
      const char* pr = pbase + (size_t)(t + 1) * (I_ * 2);
#pragma unroll
      for (int j = 0; j < 4; ++j)
        __builtin_prefetch(pr + (pcol_i + j * 4) * 128, 0, 3);
    }

    v8f acc[4];
#pragma unroll
    for (int g = 0; g < 4; ++g) {
      float iv = (kh == 0) ? bval[g] : 0.0f;
      v8f a;
#pragma unroll
      for (int r = 0; r < 8; ++r) a[r] = iv;
      acc[g] = a;
    }

    // B panel base for this wave's K-half: chunk kcg = kh*32 + kc
    unsigned bbase = ((unsigned)(kh ? 32 : 0) << 12) + ((unsigned)lane << 4);

#pragma unroll 2
    for (int kc = 0; kc < 32; ++kc) {
      v8bf alo = *(const v8bf*)(arow + kc * 32);        // K base+0..7
      v8bf ahi = *(const v8bf*)(arow + kc * 32 + 16);   // K base+16..23
      v16bf a = __builtin_shufflevector(alo, ahi,
          0, 1, 2, 3, 4, 5, 6, 7, 8, 9, 10, 11, 12, 13, 14, 15);
#pragma unroll
      for (int g = 0; g < 4; ++g) {
        // gates adjacent within chunk: offsets g*1024 (+512) fit DS immediates
        v8bf blo = *(const v8bf*)(smem + bbase + g * 1024);
        v8bf bhi = *(const v8bf*)(smem + bbase + g * 1024 + 512);
        v16bf bm = __builtin_shufflevector(blo, bhi,
            0, 1, 2, 3, 4, 5, 6, 7, 8, 9, 10, 11, 12, 13, 14, 15);
        acc[g] = __builtin_amdgcn_wmma_f32_16x16x32_bf16(
            false, a, false, bm, (short)0, acc[g], false, false);
      }
      bbase += 4096;
    }

    // ---- K-split reduction ----
    if (kh == 1) {
#pragma unroll
      for (int g = 0; g < 4; ++g)
        *(v8f*)&red[(((wave - 4) * 4 + g) * 32 + lane) * 8] = acc[g];
    }
    __syncthreads();
    if (kh == 0) {
#pragma unroll
      for (int g = 0; g < 4; ++g) {
        v8f part = *(const v8f*)&red[((wave * 4 + g) * 32 + lane) * 8];
        acc[g] = acc[g] + part;
      }
      // C layout: VGPR r -> M = r + 8*lhi, N = l16.
#pragma unroll
      for (int r = 0; r < 8; ++r) {
        int m = m0 + r + (lhi << 3);
        size_t sidx = (size_t)m * H_ + col;
        float iv = sigmoid_f(acc[0][r]);
        float fv = sigmoid_f(acc[1][r]);
        float gv = tanh_f(acc[2][r]);
        float ov = sigmoid_f(acc[3][r]);
        float c = fv * c_state[sidx] + iv * gv;
        float h = ov * tanh_f(c);
        c_state[sidx] = c;
        hout[sidx] = (__bf16)h;
        size_t oidx = ((size_t)m * T_ + t) * (size_t)H_ + col;
        hs_out[oidx] = h;
        cs_out[oidx] = c;
      }
    }

    // ---- device-wide step barrier (release h, acquire h) ----
    __syncthreads();                       // red consumed, stores issued
    if (tid == 0) {
      __threadfence();                     // release: h/c visible device-wide
      unsigned a = atomicAdd(&bar[0], 1u);
      if (a == NBLK_ - 1) {
        *(volatile unsigned*)&bar[0] = 0u;
        __threadfence();
        atomicAdd(&bar[1], 1u);
      } else {
        while (*(volatile unsigned*)&bar[1] <= (unsigned)t)
          __builtin_amdgcn_s_sleep(2);
      }
    }
    __syncthreads();
    __threadfence();                       // acquire: drop stale h lines
  }
}

extern "C" void kernel_launch(void* const* d_in, const int* in_sizes, int n_in,
                              void* d_out, int out_size, void* d_ws, size_t ws_size,
                              hipStream_t stream) {
  const float* x    = (const float*)d_in[0];
  const float* w_ih = (const float*)d_in[1];
  const float* w_hh = (const float*)d_in[2];
  const float* bias = (const float*)d_in[3];

  char* ws = (char*)d_ws;
  __bf16* wpack = (__bf16*)(ws + WPK_OFF);
  __bf16* xb    = (__bf16*)(ws + XBF_OFF);
  __bf16* h0    = (__bf16*)(ws + H0_OFF);
  __bf16* h1    = (__bf16*)(ws + H1_OFF);
  float*  cst   = (float*)(ws + CST_OFF);
  unsigned* bar = (unsigned*)(ws + BAR_OFF);

  float* hs = (float*)d_out;
  float* cs = hs + (size_t)B_ * T_ * H_;

  (void)hipFuncSetAttribute((const void*)lstm_persistent,
                            hipFuncAttributeMaxDynamicSharedMemorySize,
                            (int)LDS_BYTES_);

  pack_weights<<<(NBLK_ * 4 * 64 * 32 * 8) / 256, 256, 0, stream>>>(w_ih, w_hh, wpack);
  convert_x<<<(B_ * T_ * I_) / (256 * 4), 256, 0, stream>>>(x, xb);
  init_state<<<(B_ * H_) / 256, 256, 0, stream>>>(h0, h1, cst, bar);

  lstm_persistent<<<NBLK_, 256, LDS_BYTES_, stream>>>(
      xb, wpack, bias, h0, h1, cst, hs, cs, bar);
}
